// JointNetwork_8555574854403
// MI455X (gfx1250) — matrix operand order repaired and checked by reading, not compiled, not run on hardware
//
#include <hip/hip_runtime.h>
#include <stdint.h>

typedef __attribute__((ext_vector_type(16))) __bf16    v16bf;
typedef __attribute__((ext_vector_type(8)))  float     v8f;
typedef __attribute__((ext_vector_type(4)))  uint32_t  u32x4;

#define Bn 4
#define Tn 512
#define Un 100
#define Hn 512
#define Vn 1025
#define NTILES 65           /* 65*16 = 1040 padded columns */
#define MROWS (Bn*Tn*Un)    /* 204800 */
#define MTILE 64            /* rows per block: 4 row-groups of 16 */

union Frag {
    v16bf v;
    u32x4 q[2];
};

__device__ __forceinline__ unsigned short f2bf(float f) {
    union { float f; uint32_t u; } a; a.f = f;
    uint32_t u = a.u;
    return (unsigned short)((u + 0x7FFFu + ((u >> 16) & 1u)) >> 16);
}

// One-time: Wt[v][h] = bf16(W[h][v]), zero-padded to 1040 rows.
__global__ __launch_bounds__(256) void prep_wt(const float* __restrict__ W,
                                               unsigned short* __restrict__ Wt) {
    int v = blockIdx.x;                      // 0..1039
    for (int h = threadIdx.x; h < Hn; h += 256) {
        float x = (v < Vn) ? W[(size_t)h * Vn + v] : 0.0f;
        Wt[(size_t)v * Hn + h] = f2bf(x);
    }
}

__device__ __forceinline__ void load_a(Frag& a, const unsigned short* base, int kk) {
    const unsigned short* ap = base + kk * 32;
    a.q[0] = *(const u32x4*)(ap);            // K {0-7} or {8-15}
    a.q[1] = *(const u32x4*)(ap + 16);       // K {16-23} or {24-31}
}
__device__ __forceinline__ void load_b(Frag& b, const unsigned short* base, int kk) {
    const unsigned short* bp = base + kk * 32;
    b.q[0] = *(const u32x4*)(bp);            // K {0-7} or {8-15}
    b.q[1] = *(const u32x4*)(bp + 16);       // K {16-23} or {24-31}
}

#define WMMA_BF16(A, B, C) __builtin_amdgcn_wmma_f32_16x16x32_bf16( \
        false, (A).v, false, (B).v, (short)0, (C), false, false)

__device__ __forceinline__ void store_tile(float* __restrict__ out,
                                           const float* __restrict__ bias,
                                           int mrow0, int nt, int lrow, int lhalf,
                                           v8f acc) {
    int v = nt * 16 + lrow;
    if (v < Vn) {
        float bv = bias[v];
#pragma unroll
        for (int r = 0; r < 8; ++r) {
            int m = mrow0 + r + lhalf * 8;
            __builtin_nontemporal_store(acc[r] + bv, &out[(size_t)m * Vn + v]);
        }
    }
}

__global__ __launch_bounds__(128) void joint_wmma(
    const float* __restrict__ trans, const float* __restrict__ pred,
    const float* __restrict__ bias, const unsigned short* __restrict__ Wt,
    float* __restrict__ out)
{
    __shared__ __align__(16) unsigned short As[MTILE * Hn];   // 64 KB bf16 A-tile

    const int tid   = threadIdx.x;
    const int mbase = blockIdx.x * MTILE;

    // ---- Build A tile: tanh(trans[b,t,:] + pred[b,u,:]) in bf16 ----
    for (int r = 0; r < MTILE; ++r) {
        int gm  = mbase + r;
        int b   = gm / (Tn * Un);
        int rem = gm - b * (Tn * Un);
        int t   = rem / Un;
        int u   = rem - t * Un;
        const float* tr = trans + (size_t)(b * Tn + t) * Hn;
        const float* pr = pred  + (size_t)(b * Un + u) * Hn;
#pragma unroll
        for (int j = 0; j < Hn / 128; ++j) {
            int h = tid + j * 128;
            As[r * Hn + h] = f2bf(tanhf(tr[h] + pr[h]));
        }
    }
    __syncthreads();

    const int wave  = tid >> 5;
    const int lane  = tid & 31;
    const int lrow  = lane & 15;          // N index (B/C) or M row (A)
    const int lhalf = (lane >> 4) & 1;    // lane half selects K sub-range

    // A fragment bases for the four row-groups of this lane.
    const unsigned short* A0 = As + (lrow     ) * Hn + lhalf * 8;
    const unsigned short* A1 = As + (16 + lrow) * Hn + lhalf * 8;
    const unsigned short* A2 = As + (32 + lrow) * Hn + lhalf * 8;
    const unsigned short* A3 = As + (48 + lrow) * Hn + lhalf * 8;

    // Each wave: pairs of N-tiles (stride 8) x 4 row-groups -> 4x2 WMMA grid.
    for (int nt0 = wave * 2; nt0 < NTILES; nt0 += 8) {
        const int nt1 = nt0 + 1;
        const bool has2 = (nt1 < NTILES);   // wave-uniform

        v8f acc00 = {}, acc01 = {}, acc10 = {}, acc11 = {};
        v8f acc20 = {}, acc21 = {}, acc30 = {}, acc31 = {};
        const unsigned short* Bw0 = Wt + (size_t)(nt0 * 16 + lrow) * Hn + lhalf * 8;
        const unsigned short* Bw1 = Wt + (size_t)(nt1 * 16 + lrow) * Hn + lhalf * 8;

        if (has2) {
#pragma unroll 2
            for (int kk = 0; kk < Hn / 32; ++kk) {
                Frag a0, a1, a2, a3, b0, b1;
                load_b(b0, Bw0, kk);
                load_b(b1, Bw1, kk);
                load_a(a0, A0, kk);
                load_a(a1, A1, kk);
                load_a(a2, A2, kk);
                load_a(a3, A3, kk);
                acc00 = WMMA_BF16(a0, b0, acc00);
                acc01 = WMMA_BF16(a0, b1, acc01);
                acc10 = WMMA_BF16(a1, b0, acc10);
                acc11 = WMMA_BF16(a1, b1, acc11);
                acc20 = WMMA_BF16(a2, b0, acc20);
                acc21 = WMMA_BF16(a2, b1, acc21);
                acc30 = WMMA_BF16(a3, b0, acc30);
                acc31 = WMMA_BF16(a3, b1, acc31);
            }
        } else {
#pragma unroll 2
            for (int kk = 0; kk < Hn / 32; ++kk) {
                Frag a0, a1, a2, a3, b0;
                load_b(b0, Bw0, kk);
                load_a(a0, A0, kk);
                load_a(a1, A1, kk);
                load_a(a2, A2, kk);
                load_a(a3, A3, kk);
                acc00 = WMMA_BF16(a0, b0, acc00);
                acc10 = WMMA_BF16(a1, b0, acc10);
                acc20 = WMMA_BF16(a2, b0, acc20);
                acc30 = WMMA_BF16(a3, b0, acc30);
            }
        }

        store_tile(out, bias, mbase,      nt0, lrow, lhalf, acc00);
        store_tile(out, bias, mbase + 16, nt0, lrow, lhalf, acc10);
        store_tile(out, bias, mbase + 32, nt0, lrow, lhalf, acc20);
        store_tile(out, bias, mbase + 48, nt0, lrow, lhalf, acc30);
        if (has2) {
            store_tile(out, bias, mbase,      nt1, lrow, lhalf, acc01);
            store_tile(out, bias, mbase + 16, nt1, lrow, lhalf, acc11);
            store_tile(out, bias, mbase + 32, nt1, lrow, lhalf, acc21);
            store_tile(out, bias, mbase + 48, nt1, lrow, lhalf, acc31);
        }
    }
}

__global__ void copy_seq(const int* __restrict__ s, int* __restrict__ dst, int n) {
    int i = threadIdx.x + blockIdx.x * blockDim.x;
    if (i < n) dst[i] = s[i];
}

extern "C" void kernel_launch(void* const* d_in, const int* in_sizes, int n_in,
                              void* d_out, int out_size, void* d_ws, size_t ws_size,
                              hipStream_t stream) {
    const float* trans   = (const float*)d_in[0];
    const float* pred    = (const float*)d_in[1];
    const int*   seq_len = (const int*)d_in[2];
    const float* W       = (const float*)d_in[3];
    const float* bias    = (const float*)d_in[4];

    unsigned short* Wt = (unsigned short*)d_ws;   // 1040*512*2 = 1.04 MB

    float* out = (float*)d_out;

    // 1) transpose+convert W to bf16 (L2-resident, tiny one-time cost)
    prep_wt<<<NTILES * 16, 256, 0, stream>>>(W, Wt);

    // 2) main fused tanh + WMMA GEMM (4x2 tile grid per wave iteration)
    joint_wmma<<<MROWS / MTILE, 128, 0, stream>>>(trans, pred, bias, Wt, out);

    // 3) pass-through seq_len into the tail of d_out
    int nseq = in_sizes[2];
    int* seq_dst = (int*)d_out + ((size_t)out_size - nseq);
    copy_seq<<<1, 32, 0, stream>>>(seq_len, seq_dst, nseq);
}
